// Attention_27797028340174
// MI455X (gfx1250) — matrix operand order repaired and checked by reading, not compiled, not run on hardware
//
#include <hip/hip_runtime.h>
#include <hip/hip_bf16.h>

// ---------------------------------------------------------------------------
// Attention layer for MI455X (gfx1250, wave32, WMMA).
//   x = query[B*S, HID]; q/k/v = x@W + b; per-head flash attention; out f32.
// B=4, S=2048, H=16, D=64, HID=1024.
// FP16 WMMA (v_wmma_f32_16x16x32_f16) for all matmuls, f32 accumulation.
// Flash attention computes scores TRANSPOSED (K @ Q^T) so that K/V tiles are
// consumed in natural row-major layout and P needs no LDS relayout.
// K/V tiles are staged with double-buffered global_load_async_to_lds_b128.
// ---------------------------------------------------------------------------

#define Bc 4
#define Sc 2048
#define Hc 16
#define Dc 64
#define HIDc 1024
#define NROWS (Bc * Sc)          // 8192

typedef __attribute__((ext_vector_type(16))) _Float16 v16h;
typedef __attribute__((ext_vector_type(8)))  float    v8f;

// Build a v16h fragment from two 16-byte chunks (global or LDS).
__device__ __forceinline__ v16h load_frag2(const _Float16* p0, const _Float16* p1) {
    union { v16h h; uint4 q[2]; } u;
    u.q[0] = *(const uint4*)p0;
    u.q[1] = *(const uint4*)p1;
    return u.h;
}
// Contiguous 16 halves (32B).
__device__ __forceinline__ v16h load_frag_c(const _Float16* p) {
    return load_frag2(p, p + 8);
}

__device__ __forceinline__ v8f wmma_f16(v16h a, v16h b, v8f c) {
    return __builtin_amdgcn_wmma_f32_16x16x32_f16(
        /*neg_a=*/false, a, /*neg_b=*/false, b,
        /*c_mod=*/(short)0, c, /*reuse_a=*/false, /*reuse_b=*/false);
}

// Async copy 16B global -> LDS (CDNA5 ASYNCcnt path).
__device__ __forceinline__ void async_b128(unsigned lds_off, const void* g) {
    asm volatile("global_load_async_to_lds_b128 %0, %1, off"
                 :: "v"(lds_off), "v"(g) : "memory");
}

// ---------------------------------------------------------------------------
// f32 -> f16 conversion (grid-stride)
// ---------------------------------------------------------------------------
__global__ void cvt_f32_f16(const float* __restrict__ src,
                            _Float16* __restrict__ dst, int n) {
    int i = blockIdx.x * blockDim.x + threadIdx.x;
    int stride = gridDim.x * blockDim.x;
    for (; i < n; i += stride) dst[i] = (_Float16)src[i];
}

// ---------------------------------------------------------------------------
// QKV projection: out[b,h,s,d] = (x @ W + bias), f16 output in split-head
// layout. One wave computes a 16(M) x 64(N) tile; A fragment reused across
// 4 WMMAs per K-step.  grid = (NROWS/16, HID/64), block = 32.
//   A 16x32 f16: lane L holds row M=L%16, K chunks {8*(L/16)..+8} and +16.
//   B 32x16 f16: lane L holds col N=L%16, K chunks {8*(L/16)..+8} and +16.
// ---------------------------------------------------------------------------
__global__ void qkv_gemm(const _Float16* __restrict__ X,   // [NROWS, HID]
                         const _Float16* __restrict__ W,   // [HID, HID]
                         const float*    __restrict__ bias,// [HID]
                         _Float16*       __restrict__ out) // [B,H,S,D]
{
    const int lane = threadIdx.x & 31;
    const int m    = lane & 15;
    const int hi   = lane >> 4;
    const int m0   = blockIdx.x * 16;
    const int n0   = blockIdx.y * 64;

    const _Float16* arow = X + (size_t)(m0 + m) * HIDc;

    v8f acc[4] = {};
    for (int k0 = 0; k0 < HIDc; k0 += 32) {
        __builtin_prefetch(arow + k0 + 256, 0, 0);      // global_prefetch_b8
        const _Float16* ab = arow + k0 + 8 * hi;
        v16h a = load_frag2(ab, ab + 16);
        const _Float16* bcol = W + (size_t)(k0 + 8 * hi) * HIDc + n0 + m;
#pragma unroll
        for (int nt = 0; nt < 4; ++nt) {
            union { v16h h; _Float16 e[16]; } ub;
#pragma unroll
            for (int i = 0; i < 8; ++i) {
                ub.e[i]     = bcol[(size_t)i * HIDc + nt * 16];
                ub.e[8 + i] = bcol[(size_t)(16 + i) * HIDc + nt * 16];
            }
            acc[nt] = wmma_f16(a, ub.h, acc[nt]);
        }
    }

    // Epilogue: C/D layout: VGPR r -> row r + 8*hi, col = lane%16.
#pragma unroll
    for (int r = 0; r < 8; ++r) {
        const int grow = m0 + r + 8 * hi;          // row in [0, NROWS)
        const int bb = grow >> 11;                 // / S
        const int ss = grow & (Sc - 1);
#pragma unroll
        for (int nt = 0; nt < 4; ++nt) {
            const int col = n0 + nt * 16 + m;
            const int hh = col >> 6;               // / D
            const int dd = col & (Dc - 1);
            float v = acc[nt][r] + bias[col];
            out[(((size_t)bb * Hc + hh) * Sc + ss) * Dc + dd] = (_Float16)v;
        }
    }
}

// ---------------------------------------------------------------------------
// Flash attention per head, transposed-score formulation.
// grid = (S/64, B*H), block = 128 (4 waves); each wave owns 16 query rows.
// 32-key K/V tiles are contiguous 4KB blocks -> async-copied to LDS with
// double buffering (issue next tile, s_wait_asynccnt 4, barrier, compute).
// ---------------------------------------------------------------------------
__global__ void flash_attn(const _Float16* __restrict__ qh,  // [B,H,S,D]
                           const _Float16* __restrict__ kh,
                           const _Float16* __restrict__ vh,
                           const int*      __restrict__ msk, // [B,S,S]
                           float*          __restrict__ out) // [B*S, HID]
{
    __shared__ _Float16 kbuf[2][32 * Dc];   // raw K tiles [s][d]
    __shared__ _Float16 vbuf[2][32 * Dc];   // raw V tiles [s][d]

    const int tid  = threadIdx.x;
    const int lane = tid & 31;
    const int wave = tid >> 5;
    const int m    = lane & 15;
    const int hi   = lane >> 4;

    const int head = blockIdx.y;                    // b*H + h
    const int bb   = head >> 4;
    const int hh   = head & 15;
    const int q0   = blockIdx.x * 64 + wave * 16;   // this wave's query base

    const _Float16* qhd = qh + (size_t)head * Sc * Dc;
    const _Float16* khd = kh + (size_t)head * Sc * Dc;
    const _Float16* vhd = vh + (size_t)head * Sc * Dc;

    // Q^T B-fragments, built once, live in registers:
    // lane L holds q=q0+L%16, d chunks {8*(L/16)..+8} and +16 (bq1: +32).
    v16h bq0, bq1;
    {
        const _Float16* qr = qhd + (size_t)(q0 + m) * Dc + 8 * hi;
        bq0 = load_frag2(qr, qr + 16);
        bq1 = load_frag2(qr + 32, qr + 48);
    }

    v8f   o[4] = {};
    float m_run = -1e30f, l_run = 0.0f;

    // per-thread async staging of one K/V tile pair (4 x 16B per thread)
    const unsigned lk0 = (unsigned)(size_t)&kbuf[0][0];
    const unsigned lv0 = (unsigned)(size_t)&vbuf[0][0];
    const int tb = tid * 16;                        // byte offset 0..2047
    auto issue_tile = [&](int buf, int j0) {
        const char* gk = (const char*)(khd + (size_t)j0 * Dc);
        const char* gv = (const char*)(vhd + (size_t)j0 * Dc);
        const unsigned lk = lk0 + buf * (32 * Dc * 2);
        const unsigned lv = lv0 + buf * (32 * Dc * 2);
        async_b128(lk + tb,        gk + tb);
        async_b128(lk + tb + 2048, gk + tb + 2048);
        async_b128(lv + tb,        gv + tb);
        async_b128(lv + tb + 2048, gv + tb + 2048);
    };

    constexpr int NT = Sc / 32;                     // 64 key tiles
    issue_tile(0, 0);                               // prologue

    for (int it = 0; it < NT; ++it) {
        const int cur = it & 1;
        const int j0  = it * 32;

        __syncthreads();               // everyone done reading buf[cur^1]
        if (it + 1 < NT) {
            issue_tile(cur ^ 1, j0 + 32);
            asm volatile("s_wait_asynccnt 0x4" ::: "memory");  // tile cur done
        } else {
            asm volatile("s_wait_asynccnt 0x0" ::: "memory");
        }
        __syncthreads();               // tile cur resident for all waves

        const _Float16* kb = &kbuf[cur][0];
        const _Float16* vb = &vbuf[cur][0];

        // ---- scores^T = K_tile @ Q^T : two 16(key) x 16(q) tiles ----
        v8f s0 = {}, s1 = {};
        {
            const _Float16* a0 = kb + m * Dc + 8 * hi;          // keys 0..15
            const _Float16* a1 = kb + (16 + m) * Dc + 8 * hi;   // keys 16..31
            s0 = wmma_f16(load_frag2(a0,      a0 + 16), bq0, s0);   // d 0..31
            s0 = wmma_f16(load_frag2(a0 + 32, a0 + 48), bq1, s0);   // d 32..63
            s1 = wmma_f16(load_frag2(a1,      a1 + 16), bq0, s1);
            s1 = wmma_f16(load_frag2(a1 + 32, a1 + 48), bq1, s1);
        }

        // ---- scale + mask: element (key = j0 + kg*16 + r + 8*hi, q = q0+m)
        const size_t mrow = ((size_t)bb * Sc + (q0 + m)) * Sc + j0 + 8 * hi;
        float e0[8], e1[8];
#pragma unroll
        for (int r = 0; r < 8; ++r) {
            const float mt0 = (1.0f - (float)msk[mrow + r])      * -10000.0f;
            const float mt1 = (1.0f - (float)msk[mrow + 16 + r]) * -10000.0f;
            e0[r] = s0[r] * 0.125f + mt0;           // 1/sqrt(64)
            e1[r] = s1[r] * 0.125f + mt1;
        }

        // ---- online softmax per q column (in-lane + hi<->lo shuffle) ----
        float pm = -1e30f;
#pragma unroll
        for (int r = 0; r < 8; ++r) pm = fmaxf(pm, fmaxf(e0[r], e1[r]));
        pm = fmaxf(pm, __shfl_xor(pm, 16, 32));
        const float mnew = fmaxf(m_run, pm);
        const float corr = __expf(m_run - mnew);

        union { v16h h; _Float16 e[16]; } up;       // P in A-fragment layout!
        float ps = 0.0f;
#pragma unroll
        for (int r = 0; r < 8; ++r) {
            const float p0 = __expf(e0[r] - mnew);
            const float p1 = __expf(e1[r] - mnew);
            ps += p0 + p1;
            up.e[r]     = (_Float16)p0;             // keys 8*hi + r
            up.e[8 + r] = (_Float16)p1;             // keys 16 + 8*hi + r
        }
        ps += __shfl_xor(ps, 16, 32);
        l_run = l_run * corr + ps;
        m_run = mnew;

        // rescale accumulators: o row r holds q = r + 8*hi; stat lives in
        // lane (r + 8*hi) (and its +16 twin).
#pragma unroll
        for (int r = 0; r < 8; ++r) {
            const float cr = __shfl(corr, r + 8 * hi, 32);
#pragma unroll
            for (int g = 0; g < 4; ++g) o[g][r] *= cr;
        }

        // ---- ctx += P(16q x 32k) @ V(32k x 16d) per d-group ----
#pragma unroll
        for (int g = 0; g < 4; ++g) {
            union { v16h h; _Float16 e[16]; } ubv;
            const _Float16* vc = vb + 8 * hi * Dc + g * 16 + m;
#pragma unroll
            for (int i = 0; i < 8; ++i) {
                ubv.e[i]     = vc[i * Dc];
                ubv.e[8 + i] = vc[(16 + i) * Dc];
            }
            o[g] = wmma_f16(up.h, ubv.h, o[g]);
        }
    }

    // ---- epilogue: normalize and write f32 output [B*S, HID] ----
#pragma unroll
    for (int r = 0; r < 8; ++r) {
        const float lr  = __shfl(l_run, r + 8 * hi, 32);
        const float inv = 1.0f / lr;
        const size_t base = ((size_t)bb * Sc + (q0 + r + 8 * hi)) * HIDc + hh * Dc;
#pragma unroll
        for (int g = 0; g < 4; ++g)
            out[base + g * 16 + m] = o[g][r] * inv;
    }
}

// ---------------------------------------------------------------------------
// Host-side launch
// ---------------------------------------------------------------------------
extern "C" void kernel_launch(void* const* d_in, const int* in_sizes, int n_in,
                              void* d_out, int out_size, void* d_ws, size_t ws_size,
                              hipStream_t stream) {
    const float* query = (const float*)d_in[0];
    // d_in[1] ("key") is intentionally unused: reference re-derives K,V from query.
    const int*   mask  = (const int*)d_in[2];
    const float* Wq    = (const float*)d_in[3];
    const float* bq    = (const float*)d_in[4];
    const float* Wk    = (const float*)d_in[5];
    const float* bk    = (const float*)d_in[6];
    const float* Wv    = (const float*)d_in[7];
    const float* bv    = (const float*)d_in[8];

    const size_t NX = (size_t)NROWS * HIDc;   // 8,388,608
    const size_t NW = (size_t)HIDc * HIDc;    // 1,048,576

    _Float16* xh  = (_Float16*)d_ws;
    _Float16* wqh = xh  + NX;
    _Float16* wkh = wqh + NW;
    _Float16* wvh = wkh + NW;
    _Float16* qh  = wvh + NW;
    _Float16* kh  = qh  + NX;
    _Float16* vh  = kh  + NX;                  // total ~70 MB f16

    cvt_f32_f16<<<4096, 256, 0, stream>>>(query, xh,  (int)NX);
    cvt_f32_f16<<<1024, 256, 0, stream>>>(Wq,    wqh, (int)NW);
    cvt_f32_f16<<<1024, 256, 0, stream>>>(Wk,    wkh, (int)NW);
    cvt_f32_f16<<<1024, 256, 0, stream>>>(Wv,    wvh, (int)NW);

    dim3 ggrid(NROWS / 16, HIDc / 64);         // (512, 16)
    qkv_gemm<<<ggrid, 32, 0, stream>>>(xh, wqh, bq, qh);
    qkv_gemm<<<ggrid, 32, 0, stream>>>(xh, wkh, bk, kh);
    qkv_gemm<<<ggrid, 32, 0, stream>>>(xh, wvh, bv, vh);

    dim3 agrid(Sc / 64, Bc * Hc);              // (32, 64)
    flash_attn<<<agrid, 128, 0, stream>>>(qh, kh, vh, mask, (float*)d_out);
}
